// AttentionBlockWithSkipConnection_42116449305031
// MI455X (gfx1250) — matrix operand order repaired and checked by reading, not compiled, not run on hardware
//
#include <hip/hip_runtime.h>

// ---------------- types ----------------
typedef _Float16 v4h  __attribute__((ext_vector_type(4)));
typedef _Float16 v8h  __attribute__((ext_vector_type(8)));
typedef _Float16 v16h __attribute__((ext_vector_type(16)));
typedef float    v8f  __attribute__((ext_vector_type(8)));

union F16x16 { v16h v; v8h h[2]; };

__device__ __forceinline__ v8f wmma_f16(v16h a, v16h b, v8f c) {
  // D = A(16x32 f16) * B(32x16 f16) + C(16x16 f32)
  return __builtin_amdgcn_wmma_f32_16x16x32_f16(false, a, false, b, (short)0, c, false, false);
}

__device__ __forceinline__ v8f vzero8() {
  v8f z = {0.f,0.f,0.f,0.f,0.f,0.f,0.f,0.f};
  return z;
}

// Problem constants (B=8, H=W=64, C=256, groups=32)
#define NBATCH 8
#define NPIX   4096         // H*W
#define CH     256
#define NROW   (NBATCH*NPIX) // 32768

// ---------------- K0: weights -> f16 transposed ----------------
__global__ void k_cvt_w(const float* __restrict__ wqkv, const float* __restrict__ wproj,
                        _Float16* __restrict__ wqkvT, _Float16* __restrict__ wprojT) {
  int t = blockIdx.x * 256 + threadIdx.x;
  if (t < 768 * 256) {
    int d = t >> 8, c = t & 255;
    wqkvT[d * 256 + c] = (_Float16)wqkv[c * 768 + d];
  } else if (t < 768 * 256 + 256 * 256) {
    int t2 = t - 768 * 256;
    int d = t2 >> 8, c = t2 & 255;
    wprojT[d * 256 + c] = (_Float16)wproj[c * 256 + d];
  }
}

// ---------------- K1: GroupNorm stats ----------------
__global__ void k_gn_stats(const float* __restrict__ x,
                           float* __restrict__ mean, float* __restrict__ rstd) {
  int bg = blockIdx.x;          // b*32+g
  int b = bg >> 5, g = bg & 31;
  const float* xb = x + (size_t)b * NPIX * CH + g * 8;
  float s = 0.f, ss = 0.f;
  for (int pix = threadIdx.x; pix < NPIX; pix += 256) {
    const float4* p = (const float4*)(xb + (size_t)pix * CH);
    float4 u = p[0], v = p[1];
    s  += u.x + u.y + u.z + u.w + v.x + v.y + v.z + v.w;
    ss += u.x*u.x + u.y*u.y + u.z*u.z + u.w*u.w + v.x*v.x + v.y*v.y + v.z*v.z + v.w*v.w;
  }
  #pragma unroll
  for (int off = 16; off > 0; off >>= 1) {
    s  += __shfl_xor(s, off);
    ss += __shfl_xor(ss, off);
  }
  __shared__ float sm[8], sv[8];
  int wv = threadIdx.x >> 5, ln = threadIdx.x & 31;
  if (ln == 0) { sm[wv] = s; sv[wv] = ss; }
  __syncthreads();
  if (threadIdx.x == 0) {
    float S = 0.f, SS = 0.f;
    #pragma unroll
    for (int i = 0; i < 8; ++i) { S += sm[i]; SS += sv[i]; }
    float mu  = S * (1.f / 32768.f);
    float var = SS * (1.f / 32768.f) - mu * mu;
    mean[bg] = mu;
    rstd[bg] = rsqrtf(var + 1e-5f);
  }
}

// ---------------- K2: GN apply -> h (f16) ----------------
__global__ void k_gn_apply(const float* __restrict__ x, const float* __restrict__ mean,
                           const float* __restrict__ rstd, const float* __restrict__ scale,
                           const float* __restrict__ bias, _Float16* __restrict__ h) {
  size_t t = (size_t)blockIdx.x * 256 + threadIdx.x;  // float4 index, total 2M
  int row = (int)(t >> 6);
  int c4  = (int)(t & 63) * 4;
  int b = row >> 12, g = c4 >> 3;
  float mu = mean[b * 32 + g], rs = rstd[b * 32 + g];
  float4 u  = *(const float4*)(x + (size_t)row * CH + c4);
  float4 sc = *(const float4*)(scale + c4);
  float4 bi = *(const float4*)(bias + c4);
  v4h o;
  o.x = (_Float16)((u.x - mu) * rs * sc.x + bi.x);
  o.y = (_Float16)((u.y - mu) * rs * sc.y + bi.y);
  o.z = (_Float16)((u.z - mu) * rs * sc.z + bi.z);
  o.w = (_Float16)((u.w - mu) * rs * sc.w + bi.w);
  *(v4h*)(h + (size_t)row * CH + c4) = o;
}

// ---------------- GEMM: C[M,N] = A[M,256] * BT[N,256]^T ----------------
// block: 256 threads = 8 waves; tile 128(M) x 64(N); wave = 16 rows x 4 n-frags.
// MODE 0: qkv epilogue (write q*1/16, k, vT f16). MODE 1: proj epilogue (+bias+x, f32).
template <int MODE>
__global__ void __launch_bounds__(256, 1)
k_gemm(const _Float16* __restrict__ A, const _Float16* __restrict__ BT,
       const float* __restrict__ bias,
       _Float16* __restrict__ qf, _Float16* __restrict__ kf,
       _Float16* __restrict__ vT,
       const float* __restrict__ xres, float* __restrict__ out) {
  const int tid = threadIdx.x;
  const int wv = tid >> 5, ln = tid & 31;
  const int lr = ln & 15, hi = ln >> 4;
  const int rowA    = blockIdx.x * 128 + wv * 16 + lr;   // A-fragment row (M = ln&15)
  const int colBase = blockIdx.y * 64;

  v8f acc[4];
  #pragma unroll
  for (int nt = 0; nt < 4; ++nt) acc[nt] = vzero8();

  #pragma unroll
  for (int ks = 0; ks < 8; ++ks) {
    // A fragment: K = e + 16*hi -> contiguous 16 halves of A row
    const _Float16* pa = A + (size_t)rowA * 256 + ks * 32 + hi * 16;
    F16x16 a;
    a.h[0] = *(const v8h*)pa;
    a.h[1] = *(const v8h*)(pa + 8);
    #pragma unroll
    for (int nt = 0; nt < 4; ++nt) {
      // B fragment: N = ln&15, K = e + 16*hi -> contiguous 16 halves of BT row
      const _Float16* pb = BT + (size_t)(colBase + nt * 16 + lr) * 256 + ks * 32 + hi * 16;
      F16x16 bb;
      bb.h[0] = *(const v8h*)pb;
      bb.h[1] = *(const v8h*)(pb + 8);
      acc[nt] = wmma_f16(a.v, bb.v, acc[nt]);
    }
  }

  #pragma unroll
  for (int nt = 0; nt < 4; ++nt) {
    const int dcol = colBase + nt * 16 + lr;   // D: N = ln&15
    #pragma unroll
    for (int i = 0; i < 8; ++i) {
      const int row = blockIdx.x * 128 + wv * 16 + i + hi * 8;  // D: M = i + 8*hi
      float val = acc[nt][i];
      if (MODE == 0) {
        float wb = val + bias[dcol];
        int region = dcol >> 8;       // 0:q 1:k 2:v
        int c = dcol & 255;
        if (region == 0) {
          qf[(size_t)row * CH + c] = (_Float16)(wb * 0.0625f);  // fold 1/sqrt(256)
        } else if (region == 1) {
          kf[(size_t)row * CH + c] = (_Float16)wb;
        } else {
          int b = row >> 12, n = row & 4095;
          vT[((size_t)(b * CH + c)) * NPIX + n] = (_Float16)wb;  // store v transposed
        }
      } else {
        out[(size_t)row * CH + dcol] = val + bias[dcol] + xres[(size_t)row * CH + dcol];
      }
    }
  }
}

// ---------------- Flash attention ----------------
// grid: 8 batches * 32 q-blocks; block = 8 waves; wave owns 16 query rows, full d=256.
// launch_bounds(256,1): one block per WGP -> large VGPR budget, no spills.
// B-fragments are manually double-buffered so loads for step i+1 are in flight
// during the WMMA of step i (keeps loadcnt > 0 across the chain).
__global__ void __launch_bounds__(256, 1)
k_attn(const _Float16* __restrict__ qf, const _Float16* __restrict__ kf,
       const _Float16* __restrict__ vT, _Float16* __restrict__ of) {
  __shared__ _Float16 lds[8][16][40];   // per-wave p-transpose staging (padded stride)
  const int tid = threadIdx.x;
  const int wv = tid >> 5, ln = tid & 31;
  const int lr = ln & 15, hi = ln >> 4;
  const int b    = blockIdx.x >> 5;
  const int qblk = blockIdx.x & 31;
  const int qrow = qblk * 128 + wv * 16;

  const _Float16* qb = qf + (size_t)b * NPIX * CH;
  const _Float16* kb = kf + (size_t)b * NPIX * CH;
  const _Float16* vb = vT + (size_t)b * CH * NPIX;

  // resident q fragments: 16 rows x 256 cols = 8 K-step fragments
  v16h qfr[8];
  #pragma unroll
  for (int ks = 0; ks < 8; ++ks) {
    const _Float16* p = qb + (size_t)(qrow + lr) * 256 + ks * 32 + hi * 16;
    F16x16 a;
    a.h[0] = *(const v8h*)p;
    a.h[1] = *(const v8h*)(p + 8);
    qfr[ks] = a.v;
  }

  v8f o[16];
  #pragma unroll
  for (int ct = 0; ct < 16; ++ct) o[ct] = vzero8();
  float m[8], l[8];
  #pragma unroll
  for (int i = 0; i < 8; ++i) { m[i] = -3.0e38f; l[i] = 0.f; }

  for (int kt = 0; kt < 128; ++kt) {
    const int key0 = kt * 32;
    const _Float16* krow0 = kb + (size_t)(key0 +      lr) * 256 + hi * 16;
    const _Float16* krow1 = kb + (size_t)(key0 + 16 + lr) * 256 + hi * 16;

    // ---- S = q * k^T, double-buffered k fragments ----
    v8f s0 = vzero8(), s1 = vzero8();
    F16x16 kb0[2], kb1[2];
    kb0[0].h[0] = *(const v8h*)krow0;
    kb0[0].h[1] = *(const v8h*)(krow0 + 8);
    kb1[0].h[0] = *(const v8h*)krow1;
    kb1[0].h[1] = *(const v8h*)(krow1 + 8);
    #pragma unroll
    for (int ks = 0; ks < 8; ++ks) {
      const int cur = ks & 1, nxt = cur ^ 1;
      if (ks < 7) {
        const _Float16* p0 = krow0 + (ks + 1) * 32;
        const _Float16* p1 = krow1 + (ks + 1) * 32;
        kb0[nxt].h[0] = *(const v8h*)p0;
        kb0[nxt].h[1] = *(const v8h*)(p0 + 8);
        kb1[nxt].h[0] = *(const v8h*)p1;
        kb1[nxt].h[1] = *(const v8h*)(p1 + 8);
      }
      s0 = wmma_f16(qfr[ks], kb0[cur].v, s0);
      s1 = wmma_f16(qfr[ks], kb1[cur].v, s1);
    }

    // ---- online softmax per row (rows 0-7 in lanes 0-15, rows 8-15 in 16-31) ----
    v8f corrv;
    #pragma unroll
    for (int i = 0; i < 8; ++i) {
      float v0 = s0[i], v1 = s1[i];
      float mx = fmaxf(v0, v1);
      mx = fmaxf(mx, __shfl_xor(mx, 1));
      mx = fmaxf(mx, __shfl_xor(mx, 2));
      mx = fmaxf(mx, __shfl_xor(mx, 4));
      mx = fmaxf(mx, __shfl_xor(mx, 8));
      float mn   = fmaxf(m[i], mx);
      float corr = __expf(m[i] - mn);
      float p0 = __expf(v0 - mn);
      float p1 = __expf(v1 - mn);
      float rs = p0 + p1;
      rs += __shfl_xor(rs, 1);
      rs += __shfl_xor(rs, 2);
      rs += __shfl_xor(rs, 4);
      rs += __shfl_xor(rs, 8);
      l[i] = l[i] * corr + rs;
      m[i] = mn;
      corrv[i] = corr;
      int row = i + hi * 8;
      lds[wv][row][lr]      = (_Float16)p0;
      lds[wv][row][16 + lr] = (_Float16)p1;
    }
    #pragma unroll
    for (int ct = 0; ct < 16; ++ct) o[ct] *= corrv;

    // re-read p in A-fragment layout (same-wave DS is in-order; no barrier needed)
    F16x16 ap;
    ap.h[0] = *(const v8h*)&lds[wv][lr][hi * 16];
    ap.h[1] = *(const v8h*)&lds[wv][lr][hi * 16 + 8];

    // ---- o += p * v, double-buffered v fragments (v stored transposed) ----
    const _Float16* vrow = vb + (size_t)lr * NPIX + key0 + hi * 16;
    F16x16 bv[2];
    bv[0].h[0] = *(const v8h*)vrow;
    bv[0].h[1] = *(const v8h*)(vrow + 8);
    #pragma unroll
    for (int ct = 0; ct < 16; ++ct) {
      const int cur = ct & 1, nxt = cur ^ 1;
      if (ct < 15) {
        const _Float16* pv = vrow + (size_t)(ct + 1) * 16 * NPIX;
        bv[nxt].h[0] = *(const v8h*)pv;
        bv[nxt].h[1] = *(const v8h*)(pv + 8);
      }
      o[ct] = wmma_f16(ap.v, bv[cur].v, o[ct]);
    }
  }

  // normalize and store f16
  #pragma unroll
  for (int i = 0; i < 8; ++i) {
    float inv = 1.f / l[i];
    int row = qrow + i + hi * 8;
    #pragma unroll
    for (int ct = 0; ct < 16; ++ct) {
      of[((size_t)b * NPIX + row) * CH + ct * 16 + lr] = (_Float16)(o[ct][i] * inv);
    }
  }
}

// ---------------- host launcher ----------------
extern "C" void kernel_launch(void* const* d_in, const int* in_sizes, int n_in,
                              void* d_out, int out_size, void* d_ws, size_t ws_size,
                              hipStream_t stream) {
  (void)in_sizes; (void)n_in; (void)out_size; (void)ws_size;
  const float* x        = (const float*)d_in[0];
  const float* gn_scale = (const float*)d_in[1];
  const float* gn_bias  = (const float*)d_in[2];
  const float* w_qkv    = (const float*)d_in[3];
  const float* b_qkv    = (const float*)d_in[4];
  const float* w_proj   = (const float*)d_in[5];
  const float* b_proj   = (const float*)d_in[6];

  char* ws = (char*)d_ws;
  float*    mean   = (float*)ws;                              // 256 f32
  float*    rstd   = mean + 256;                              // 256 f32
  _Float16* wqkvT  = (_Float16*)(ws + 2048);                  // 768*256 f16
  _Float16* wprojT = (_Float16*)(ws + 2048 + 768 * 256 * 2);  // 256*256 f16
  _Float16* h      = (_Float16*)(ws + 2048 + 768 * 256 * 2 + 256 * 256 * 2);
  _Float16* q      = h + (size_t)NROW * CH;
  _Float16* k      = q + (size_t)NROW * CH;
  _Float16* vT     = k + (size_t)NROW * CH;
  _Float16* of     = h;   // h no longer needed after QKV GEMM; reuse for attention output

  k_cvt_w<<<1024, 256, 0, stream>>>(w_qkv, w_proj, wqkvT, wprojT);
  k_gn_stats<<<256, 256, 0, stream>>>(x, mean, rstd);
  k_gn_apply<<<8192, 256, 0, stream>>>(x, mean, rstd, gn_scale, gn_bias, h);
  k_gemm<0><<<dim3(NROW / 128, 768 / 64), 256, 0, stream>>>(
      h, wqkvT, b_qkv, q, k, vT, nullptr, nullptr);
  k_attn<<<NBATCH * 32, 256, 0, stream>>>(q, k, vT, of);
  k_gemm<1><<<dim3(NROW / 128, 256 / 64), 256, 0, stream>>>(
      of, wprojT, b_proj, nullptr, nullptr, nullptr, x, (float*)d_out);
}
